// STABlock_80066780332297
// MI455X (gfx1250) — compile-verified, matrix operand order checked by reading, not run on hardware
//
#include <hip/hip_runtime.h>
#include <hip/hip_bf16.h>
#include <math.h>

#define B_   8
#define N_   2048
#define F_   16
#define T_   12
#define K_   3
#define FC_  64
#define FT_  64
#define CDIM 192   // F_*T_

typedef __attribute__((ext_vector_type(16))) __bf16 v16bf;
typedef __attribute__((ext_vector_type(8)))  float  v8f;
typedef __attribute__((ext_vector_type(8)))  unsigned short u16x8;

union Frag { unsigned u[8]; v16bf v; };
union Acc  { float f[8]; v8f v; };

__device__ __forceinline__ unsigned short f2bf(float f) {
    union { float f; unsigned u; } v; v.f = f;
    unsigned r = v.u + 0x7FFFu + ((v.u >> 16) & 1u);   // round-to-nearest-even
    return (unsigned short)(r >> 16);
}
__device__ __forceinline__ float sigm(float x) { return 1.0f / (1.0f + expf(-x)); }

// K-pair base for the 16x16x32 bf16 A/B fragment layout:
// lanes 0-15 hold K in {0..7} u {16..23}; lanes 16-31 hold {8..15} u {24..31}
__device__ __forceinline__ int kpair(int v, int khalf) {
    return (v < 4 ? 2 * v : 16 + 2 * (v - 4)) + 8 * khalf;
}

// ---------------- temporal attention ----------------

// tmp1[b,t,f] = sum_n x[b,n,f,t] * U1[n]
__global__ void tmp1_kernel(const float* __restrict__ x, const float* __restrict__ U1,
                            float* __restrict__ tmp1) {
    int b = blockIdx.x / T_, t = blockIdx.x % T_;
    int f = threadIdx.x & 15, g = threadIdx.x >> 4;
    float s = 0.f;
    for (int n = g; n < N_; n += 16)
        s += x[(((size_t)b * N_ + n) * F_ + f) * T_ + t] * U1[n];
    __shared__ float red[16][17];
    red[g][f] = s;
    __syncthreads();
    if (threadIdx.x < 16) {
        float acc = 0.f;
        for (int gg = 0; gg < 16; ++gg) acc += red[gg][threadIdx.x];
        tmp1[(b * T_ + t) * F_ + threadIdx.x] = acc;
    }
}

// lhs[b,t,n] = sum_f tmp1[b,t,f] * U2[f,n]
__global__ void lhs_kernel(const float* __restrict__ tmp1, const float* __restrict__ U2,
                           float* __restrict__ lhs) {
    int idx = blockIdx.x * 256 + threadIdx.x;
    if (idx >= B_ * T_ * N_) return;
    int n = idx % N_, bt = idx / N_;
    float s = 0.f;
    for (int f = 0; f < F_; ++f) s += tmp1[bt * F_ + f] * U2[f * N_ + n];
    lhs[idx] = s;
}

// rhs[b,n,t] = sum_f U3[f] * x[b,n,f,t]
__global__ void rhs_kernel(const float* __restrict__ x, const float* __restrict__ U3,
                           float* __restrict__ rhs) {
    int idx = blockIdx.x * 256 + threadIdx.x;
    if (idx >= B_ * N_ * T_) return;
    int t = idx % T_;
    size_t bn = (size_t)(idx / T_);
    float s = 0.f;
    for (int f = 0; f < F_; ++f) s += U3[f] * x[(bn * F_ + f) * T_ + t];
    rhs[idx] = s;
}

// eprod[b,i,j] = sum_n lhs[b,i,n] * rhs[b,n,j]
__global__ void eprod_kernel(const float* __restrict__ lhs, const float* __restrict__ rhs,
                             float* __restrict__ eprod) {
    int o = blockIdx.x;
    int b = o / (T_ * T_), r = o % (T_ * T_), i = r / T_, j = r % T_;
    float s = 0.f;
    for (int n = threadIdx.x; n < N_; n += 256)
        s += lhs[(b * T_ + i) * N_ + n] * rhs[((size_t)b * N_ + n) * T_ + j];
    __shared__ float red[256];
    red[threadIdx.x] = s;
    __syncthreads();
    for (int st = 128; st > 0; st >>= 1) {
        if (threadIdx.x < st) red[threadIdx.x] += red[threadIdx.x + st];
        __syncthreads();
    }
    if (threadIdx.x == 0) eprod[o] = red[0];
}

// E = softmax_axis1( Ve @ sigmoid(eprod + be) ), one block per batch
__global__ void temporal_E_kernel(const float* __restrict__ eprod, const float* __restrict__ be,
                                  const float* __restrict__ Ve, float* __restrict__ E) {
    __shared__ float sg[T_][T_], e1[T_][T_];
    int b = blockIdx.x, tid = threadIdx.x;
    if (tid < T_ * T_) {
        int i = tid / T_, j = tid % T_;
        sg[i][j] = sigm(eprod[(b * T_ + i) * T_ + j] + be[i * T_ + j]);
    }
    __syncthreads();
    if (tid < T_ * T_) {
        int i = tid / T_, j = tid % T_;
        float s = 0.f;
        for (int k = 0; k < T_; ++k) s += Ve[i * T_ + k] * sg[k][j];
        e1[i][j] = s;
    }
    __syncthreads();
    if (tid < T_) {
        int j = tid;
        float mx = -3.0e38f;
        for (int i = 0; i < T_; ++i) mx = fmaxf(mx, e1[i][j]);
        float sum = 0.f;
        for (int i = 0; i < T_; ++i) sum += expf(e1[i][j] - mx);
        float inv = 1.0f / sum;
        for (int i = 0; i < T_; ++i)
            E[(b * T_ + i) * T_ + j] = expf(e1[i][j] - mx) * inv;
    }
}

// xtat[b,n,f,t] = sum_s x[b,n,f,s] * E[b,s,t]
__global__ void xtat_kernel(const float* __restrict__ x, const float* __restrict__ E,
                            float* __restrict__ xtat) {
    int idx = blockIdx.x * 256 + threadIdx.x;
    if (idx >= B_ * N_ * F_ * T_) return;
    int t = idx % T_;
    size_t bnf = (size_t)(idx / T_);
    int b = idx / (N_ * F_ * T_);
    float s = 0.f;
    for (int ss = 0; ss < T_; ++ss)
        s += x[bnf * T_ + ss] * E[(b * T_ + ss) * T_ + t];
    xtat[idx] = s;
}

// ---------------- spatial attention ----------------

// tmp2[b,n,f] = sum_t xtat[b,n,f,t] * W1[t]
__global__ void tmp2_kernel(const float* __restrict__ xtat, const float* __restrict__ W1,
                            float* __restrict__ tmp2) {
    int idx = blockIdx.x * 256 + threadIdx.x;
    if (idx >= B_ * N_ * F_) return;
    float s = 0.f;
    for (int t = 0; t < T_; ++t) s += xtat[(size_t)idx * T_ + t] * W1[t];
    tmp2[idx] = s;
}

// sl[b,n,t] = sum_f tmp2[b,n,f] * W2[f,t]
__global__ void sl_kernel(const float* __restrict__ tmp2, const float* __restrict__ W2,
                          float* __restrict__ sl) {
    int idx = blockIdx.x * 256 + threadIdx.x;
    if (idx >= B_ * N_ * T_) return;
    int t = idx % T_, bn = idx / T_;
    float s = 0.f;
    for (int f = 0; f < F_; ++f) s += tmp2[bn * F_ + f] * W2[f * T_ + t];
    sl[idx] = s;
}

// sr[b,t,n] = sum_f W3[f] * xtat[b,n,f,t]
__global__ void sr_kernel(const float* __restrict__ xtat, const float* __restrict__ W3,
                          float* __restrict__ sr) {
    int idx = blockIdx.x * 256 + threadIdx.x;
    if (idx >= B_ * T_ * N_) return;
    int n = idx % N_, t = (idx / N_) % T_, b = idx / (N_ * T_);
    float s = 0.f;
    for (int f = 0; f < F_; ++f)
        s += W3[f] * xtat[(((size_t)b * N_ + n) * F_ + f) * T_ + t];
    sr[idx] = s;
}

// sig[b,i,j] = bf16( sigmoid(sum_t sl[b,i,t]*sr[b,t,j] + bs[i,j]) )   (bf16 out: halves HBM traffic)
__global__ void spre_kernel(const float* __restrict__ sl, const float* __restrict__ sr,
                            const float* __restrict__ bs, unsigned short* __restrict__ sig) {
    size_t idx = (size_t)blockIdx.x * 256 + threadIdx.x;
    int j = (int)(idx % N_);
    size_t bi = idx / N_;
    int i = (int)(bi % N_), b = (int)(bi / N_);
    float s = bs[(size_t)i * N_ + j];
    for (int t = 0; t < T_; ++t)
        s += sl[((size_t)b * N_ + i) * T_ + t] * sr[((size_t)b * T_ + t) * N_ + j];
    sig[idx] = f2bf(sigm(s));
}

// one-time f32 -> bf16 conversion (used for Vs)
__global__ void cvt_bf16_kernel(const float* __restrict__ src, unsigned short* __restrict__ dst,
                                int nElem) {
    int idx = blockIdx.x * 256 + threadIdx.x;
    if (idx < nElem) dst[idx] = f2bf(src[idx]);
}

// ---------------- WMMA GEMM:  S[b] = Vs (NxN, bf16) @ sig[b] (NxN, bf16) ----------------
// A tile: async-copied global->LDS (double buffered, overlapped with WMMA).
// B tile: register-prefetched one tile ahead, transposed into LDS ([n][k] so K-pairs are dwords).
#define GSTEPS (N_ / 32)
__global__ __launch_bounds__(256) void gemm_vs_kernel(const unsigned short* __restrict__ Avs,
                                                      const unsigned short* __restrict__ sig,
                                                      float* __restrict__ S) {
    __shared__ alignas(16) unsigned short Alds[2][128][40];  // [m][k], 80B row stride (16B chunks stay aligned)
    __shared__ alignas(16) unsigned short Blds[128][40];     // B^T: [n][k]
    int b = blockIdx.z;
    const unsigned short* Bg = sig + (size_t)b * N_ * N_;
    float* Cg = S + (size_t)b * N_ * N_;
    int m0 = blockIdx.y * 128, n0 = blockIdx.x * 128;
    int tid = threadIdx.x, wave = tid >> 5, lane = tid & 31;
    int wm = wave >> 1, wn = wave & 1;  // wave: 32 rows (2 M-tiles) x 64 cols (4 N-tiles)
    Acc acc[2][4];
    for (int i = 0; i < 2; ++i)
        for (int j = 0; j < 4; ++j)
            for (int r = 0; r < 8; ++r) acc[i][j].f[r] = 0.f;

    // A async staging: 512 16B-chunks per tile, 2 per thread (rows ar, ar+64; quarter aq)
    const int ar = tid >> 2, aq = tid & 3;
    // B prefetch: 2 u16x8 per thread (rows br, br+16; 8-col chunk bc8)
    const int br = tid >> 4, bc8 = (tid & 15) * 8;

    auto issueA = [&](int kb, int buf) {
        unsigned l0 = (unsigned)(size_t)&Alds[buf][ar][aq * 8];
        unsigned long long g0 =
            (unsigned long long)(Avs + (size_t)(m0 + ar) * N_ + kb + aq * 8);
        asm volatile("global_load_async_to_lds_b128 %0, %1, off" :: "v"(l0), "v"(g0) : "memory");
        unsigned l1 = (unsigned)(size_t)&Alds[buf][ar + 64][aq * 8];
        unsigned long long g1 = g0 + (unsigned long long)(64 * N_ * 2);
        asm volatile("global_load_async_to_lds_b128 %0, %1, off" :: "v"(l1), "v"(g1) : "memory");
    };
    auto loadB = [&](int kb, u16x8& v0, u16x8& v1) {
        v0 = *(const u16x8*)(Bg + (size_t)(kb + br) * N_ + n0 + bc8);
        v1 = *(const u16x8*)(Bg + (size_t)(kb + br + 16) * N_ + n0 + bc8);
    };

    u16x8 pb0, pb1;
    issueA(0, 0);
    loadB(0, pb0, pb1);

    for (int step = 0; step < GSTEPS; ++step) {
        __syncthreads();  // everyone done reading Blds and Alds[step&1] from prev iter
        // commit prefetched B tile (transposed)
        for (int e = 0; e < 8; ++e) {
            Blds[bc8 + e][br]      = pb0[e];
            Blds[bc8 + e][br + 16] = pb1[e];
        }
        if (step + 1 < GSTEPS) {
            issueA((step + 1) * 32, (step + 1) & 1);       // overlaps with this step's WMMAs
            loadB((step + 1) * 32, pb0, pb1);
            asm volatile("s_wait_asynccnt 0x2" ::: "memory");  // tile `step` done; tile step+1 in flight
        } else {
            asm volatile("s_wait_asynccnt 0x0" ::: "memory");
        }
        __syncthreads();  // all waves' async chunks + B stores visible

        const unsigned short (*Ab)[40] = Alds[step & 1];
        int mrow = lane & 15, khalf = lane >> 4;
        Frag af[2], bf[4];
        for (int v = 0; v < 8; ++v) {
            int k2 = kpair(v, khalf);
            af[0].u[v] = *(const unsigned*)&Ab[wm * 32 + mrow][k2];
            af[1].u[v] = *(const unsigned*)&Ab[wm * 32 + 16 + mrow][k2];
            for (int j = 0; j < 4; ++j)
                bf[j].u[v] = *(const unsigned*)&Blds[wn * 64 + j * 16 + mrow][k2];
        }
        for (int i = 0; i < 2; ++i)
            for (int j = 0; j < 4; ++j)
                acc[i][j].v = __builtin_amdgcn_wmma_f32_16x16x32_bf16(
                    false, af[i].v, false, bf[j].v, (short)0, acc[i][j].v, false, false);
    }
    int ncol = lane & 15, mhi = lane >> 4;
    for (int i = 0; i < 2; ++i)
        for (int j = 0; j < 4; ++j)
            for (int r = 0; r < 8; ++r) {
                int mm = m0 + wm * 32 + i * 16 + r + 8 * mhi;
                int nn = n0 + wn * 64 + j * 16 + ncol;
                Cg[(size_t)mm * N_ + nn] = acc[i][j].f[r];
            }
}

// column softmax over axis=1 of S[b,i,j], in place
__global__ void softmax_col_kernel(float* __restrict__ S) {
    int b = blockIdx.y;
    int jl = threadIdx.x & 31, ig = threadIdx.x >> 5;
    int j = blockIdx.x * 32 + jl;
    float* base = S + (size_t)b * N_ * N_;
    __shared__ float red[8][33];
    __shared__ float colv[32];
    float m = -3.0e38f;
    for (int i = ig; i < N_; i += 8) m = fmaxf(m, base[(size_t)i * N_ + j]);
    red[ig][jl] = m;
    __syncthreads();
    if (ig == 0) {
        float mm = red[0][jl];
        for (int g = 1; g < 8; ++g) mm = fmaxf(mm, red[g][jl]);
        colv[jl] = mm;
    }
    __syncthreads();
    float mx = colv[jl];
    float s = 0.f;
    for (int i = ig; i < N_; i += 8) s += expf(base[(size_t)i * N_ + j] - mx);
    __syncthreads();
    red[ig][jl] = s;
    __syncthreads();
    if (ig == 0) {
        float ss = 0.f;
        for (int g = 0; g < 8; ++g) ss += red[g][jl];
        colv[jl] = 1.0f / ss;
    }
    __syncthreads();
    float inv = colv[jl];
    for (int i = ig; i < N_; i += 8) {
        size_t o = (size_t)i * N_ + j;
        base[o] = expf(base[o] - mx) * inv;
    }
}

// ---------------- WMMA GEMM: rhsk[k,b] = (cheb_k * A_b)^T @ X_b  (NxN @ NxCDIM) ----------------
__global__ __launch_bounds__(256) void cheb_gemm_kernel(const float* __restrict__ cheb,
                                                        const float* __restrict__ A,
                                                        const float* __restrict__ x,
                                                        float* __restrict__ rhsk) {
    __shared__ alignas(16) unsigned short At[128][34];  // M^T: [j][i]
    __shared__ alignas(16) unsigned short Bt[64][34];   // X^T: [c][i]
    int kz = blockIdx.z;
    int kk = kz / B_, b = kz % B_;
    const float* ch = cheb + (size_t)kk * N_ * N_;
    const float* Ab = A + (size_t)b * N_ * N_;
    const float* Xb = x + (size_t)b * N_ * CDIM;
    float* out = rhsk + (size_t)kz * N_ * CDIM;
    int j0 = blockIdx.y * 128, c0 = blockIdx.x * 64;
    int tid = threadIdx.x, wave = tid >> 5, lane = tid & 31;
    int wj = wave >> 1, wc = wave & 1;  // wave: 32 rows x 32 cols (2x2 tiles)
    Acc acc[2][2];
    for (int i = 0; i < 2; ++i)
        for (int j = 0; j < 2; ++j)
            for (int r = 0; r < 8; ++r) acc[i][j].f[r] = 0.f;

    for (int ib = 0; ib < N_; ib += 32) {
        __syncthreads();
        for (int idx = tid; idx < 1024; idx += 256) {        // 32x128 tile, float4 chunks
            int r = idx >> 5, c4 = (idx & 31) * 4;
            size_t g = (size_t)(ib + r) * N_ + j0 + c4;
            float4 ca = *(const float4*)(ch + g);
            float4 aa = *(const float4*)(Ab + g);
            At[c4 + 0][r] = f2bf(ca.x * aa.x);               // fused cheb*A, transposed
            At[c4 + 1][r] = f2bf(ca.y * aa.y);
            At[c4 + 2][r] = f2bf(ca.z * aa.z);
            At[c4 + 3][r] = f2bf(ca.w * aa.w);
        }
        for (int idx = tid; idx < 512; idx += 256) {         // 32x64 tile, float4 chunks
            int r = idx >> 4, c4 = (idx & 15) * 4;
            float4 xv = *(const float4*)(Xb + (size_t)(ib + r) * CDIM + c0 + c4);
            Bt[c4 + 0][r] = f2bf(xv.x);
            Bt[c4 + 1][r] = f2bf(xv.y);
            Bt[c4 + 2][r] = f2bf(xv.z);
            Bt[c4 + 3][r] = f2bf(xv.w);
        }
        __syncthreads();
        int mrow = lane & 15, khalf = lane >> 4;
        Frag af[2], bf[2];
        for (int v = 0; v < 8; ++v) {
            int k2 = kpair(v, khalf);
            af[0].u[v] = *(const unsigned*)&At[wj * 32 + mrow][k2];
            af[1].u[v] = *(const unsigned*)&At[wj * 32 + 16 + mrow][k2];
            bf[0].u[v] = *(const unsigned*)&Bt[wc * 32 + mrow][k2];
            bf[1].u[v] = *(const unsigned*)&Bt[wc * 32 + 16 + mrow][k2];
        }
        for (int i = 0; i < 2; ++i)
            for (int j = 0; j < 2; ++j)
                acc[i][j].v = __builtin_amdgcn_wmma_f32_16x16x32_bf16(
                    false, af[i].v, false, bf[j].v, (short)0, acc[i][j].v, false, false);
    }
    int ncol = lane & 15, mhi = lane >> 4;
    for (int i = 0; i < 2; ++i)
        for (int j = 0; j < 2; ++j)
            for (int r = 0; r < 8; ++r) {
                int row = j0 + wj * 32 + i * 16 + r + 8 * mhi;
                int col = c0 + wc * 32 + j * 16 + ncol;
                out[(size_t)row * CDIM + col] = acc[i][j].f[r];
            }
}

// gcn[b,j,o,t] = relu( sum_k sum_f rhsk[k,b,j,f,t] * Theta[k,f,o] )
__global__ void gcn_kernel(const float* __restrict__ rhsk, const float* __restrict__ Theta,
                           float* __restrict__ gcn) {
    int idx = blockIdx.x * 256 + threadIdx.x;
    if (idx >= B_ * N_ * FC_ * T_) return;
    int t = idx % T_;
    int o = (idx / T_) % FC_;
    int bj = idx / (T_ * FC_);
    int b = bj / N_, j = bj % N_;
    float s = 0.f;
    for (int k = 0; k < K_; ++k) {
        size_t base = (((size_t)(k * B_ + b)) * N_ + j) * CDIM;
        for (int f = 0; f < F_; ++f)
            s += rhsk[base + f * T_ + t] * Theta[(k * F_ + f) * FC_ + o];
    }
    gcn[idx] = fmaxf(s, 0.f);
}

// fused: time conv(1,3) + residual 1x1 + ReLU + LayerNorm(FT) -> out (B,N,FT,T)
__global__ __launch_bounds__(256) void final_kernel(const float* __restrict__ gcn,
                                                    const float* __restrict__ x,
                                                    const float* __restrict__ tcw,
                                                    const float* __restrict__ tcb,
                                                    const float* __restrict__ rcw,
                                                    const float* __restrict__ rcb,
                                                    const float* __restrict__ lng,
                                                    const float* __restrict__ lnb,
                                                    float* __restrict__ outp) {
    __shared__ float g_s[FC_][T_];
    __shared__ float x_s[F_][T_];
    __shared__ float w_s[FT_ * FC_ * 3];
    __shared__ float rw_s[FT_ * F_];
    __shared__ float r1[4][FT_], r2[4][FT_];
    size_t bn = blockIdx.x;
    int tid = threadIdx.x, o = tid & 63, ts = tid >> 6;
    const float* gb = gcn + bn * FC_ * T_;
    const float* xb = x + bn * F_ * T_;
    for (int i = tid; i < FC_ * T_; i += 256) g_s[i / T_][i % T_] = gb[i];
    for (int i = tid; i < F_ * T_; i += 256) x_s[i / T_][i % T_] = xb[i];
    for (int i = tid; i < FT_ * FC_ * 3; i += 256) w_s[i] = tcw[i];
    for (int i = tid; i < FT_ * F_; i += 256) rw_s[i] = rcw[i];
    __syncthreads();
    float tb = tcb[o], rb = rcb[o], g = lng[o], bb = lnb[o];
    for (int tl = 0; tl < 3; ++tl) {
        int t = tl * 4 + ts;
        float s = tb;
        for (int c = 0; c < FC_; ++c) {
            const float* w = &w_s[(o * FC_ + c) * 3];
            if (t > 0)      s += g_s[c][t - 1] * w[0];
                            s += g_s[c][t]     * w[1];
            if (t < T_ - 1) s += g_s[c][t + 1] * w[2];
        }
        float r = rb;
        for (int f = 0; f < F_; ++f) r += x_s[f][t] * rw_s[o * F_ + f];
        float pre = fmaxf(s + r, 0.f);
        r1[ts][o] = pre;
        r2[ts][o] = pre * pre;
        __syncthreads();
        for (int st = 32; st >= 1; st >>= 1) {
            if (o < st) { r1[ts][o] += r1[ts][o + st]; r2[ts][o] += r2[ts][o + st]; }
            __syncthreads();
        }
        float mu = r1[ts][0] * (1.0f / FT_);
        float var = r2[ts][0] * (1.0f / FT_) - mu * mu;
        float nv = (pre - mu) * rsqrtf(var + 1e-5f);
        outp[(bn * FT_ + o) * T_ + t] = nv * g + bb;
        __syncthreads();
    }
}

extern "C" void kernel_launch(void* const* d_in, const int* in_sizes, int n_in,
                              void* d_out, int out_size, void* d_ws, size_t ws_size,
                              hipStream_t stream) {
    (void)in_sizes; (void)n_in; (void)out_size; (void)ws_size;
    const float* x     = (const float*)d_in[0];
    const float* cheb  = (const float*)d_in[1];
    const float* U1    = (const float*)d_in[2];
    const float* U2    = (const float*)d_in[3];
    const float* U3    = (const float*)d_in[4];
    const float* be    = (const float*)d_in[5];
    const float* Ve    = (const float*)d_in[6];
    const float* W1    = (const float*)d_in[7];
    const float* W2    = (const float*)d_in[8];
    const float* W3    = (const float*)d_in[9];
    const float* bs    = (const float*)d_in[10];
    const float* Vs    = (const float*)d_in[11];
    const float* Theta = (const float*)d_in[12];
    const float* tcw   = (const float*)d_in[13];
    const float* tcb   = (const float*)d_in[14];
    const float* rcw   = (const float*)d_in[15];
    const float* rcb   = (const float*)d_in[16];
    const float* lng   = (const float*)d_in[17];
    const float* lnb   = (const float*)d_in[18];
    float* outp = (float*)d_out;
    float* wsf = (float*)d_ws;

    const size_t NN = (size_t)N_ * N_;
    unsigned short* SIG = (unsigned short*)wsf;          // B*NN bf16 (sigmoid output)
    float* SA    = wsf + (size_t)B_ * NN / 2;            // B*NN f32 (S, then A in place)
    float* XTAT  = SA + (size_t)B_ * NN;                 // B*N*F*T
    float* LHS   = XTAT + (size_t)B_ * N_ * F_ * T_;     // B*T*N
    float* RHS   = LHS + (size_t)B_ * T_ * N_;           // B*N*T
    float* SL    = RHS + (size_t)B_ * N_ * T_;           // B*N*T
    float* SR    = SL + (size_t)B_ * N_ * T_;            // B*T*N
    float* TMP1  = SR + (size_t)B_ * T_ * N_;            // B*T*F
    float* TMP2  = TMP1 + (size_t)B_ * T_ * F_;          // B*N*F
    float* EPROD = TMP2 + (size_t)B_ * N_ * F_;          // B*T*T
    float* EMAT  = EPROD + (size_t)B_ * T_ * T_;         // B*T*T
    unsigned short* VSB = (unsigned short*)(EMAT + (size_t)B_ * T_ * T_);  // NN bf16 (Vs)
    // region reuse: SIG dead after gemm_vs; A (in SA) dead after cheb_gemm
    float* RHSK = (float*)SIG;                           // K*B*N*CDIM = 9.44M floats <= 16.78M
    float* GCN  = SA;                                    // B*N*FC*T = 12.58M floats <= 33.55M

    // ---- temporal attention ----
    tmp1_kernel<<<B_ * T_, 256, 0, stream>>>(x, U1, TMP1);
    rhs_kernel<<<(B_ * N_ * T_ + 255) / 256, 256, 0, stream>>>(x, U3, RHS);
    cvt_bf16_kernel<<<(int)(NN / 256), 256, 0, stream>>>(Vs, VSB, (int)NN);
    lhs_kernel<<<(B_ * T_ * N_ + 255) / 256, 256, 0, stream>>>(TMP1, U2, LHS);
    eprod_kernel<<<B_ * T_ * T_, 256, 0, stream>>>(LHS, RHS, EPROD);
    temporal_E_kernel<<<B_, 192, 0, stream>>>(EPROD, be, Ve, EMAT);
    xtat_kernel<<<(B_ * N_ * F_ * T_ + 255) / 256, 256, 0, stream>>>(x, EMAT, XTAT);

    // ---- spatial attention ----
    tmp2_kernel<<<(B_ * N_ * F_ + 255) / 256, 256, 0, stream>>>(XTAT, W1, TMP2);
    sl_kernel<<<(B_ * N_ * T_ + 255) / 256, 256, 0, stream>>>(TMP2, W2, SL);
    sr_kernel<<<(B_ * T_ * N_ + 255) / 256, 256, 0, stream>>>(XTAT, W3, SR);
    spre_kernel<<<(unsigned)((size_t)B_ * NN / 256), 256, 0, stream>>>(SL, SR, bs, SIG);
    gemm_vs_kernel<<<dim3(N_ / 128, N_ / 128, B_), 256, 0, stream>>>(VSB, SIG, SA);
    softmax_col_kernel<<<dim3(N_ / 32, B_), 256, 0, stream>>>(SA);

    // ---- chebyshev graph conv ----
    cheb_gemm_kernel<<<dim3(CDIM / 64, N_ / 128, K_ * B_), 256, 0, stream>>>(cheb, SA, x, RHSK);
    gcn_kernel<<<(B_ * N_ * FC_ * T_ + 255) / 256, 256, 0, stream>>>(RHSK, Theta, GCN);

    // ---- time conv + residual + relu + layernorm ----
    final_kernel<<<B_ * N_, 256, 0, stream>>>(GCN, x, tcw, tcb, rcw, rcb, lng, lnb, outp);
}
